// BinaryTreeLSTM_10179072491855
// MI455X (gfx1250) — matrix operand order repaired and checked by reading, not compiled, not run on hardware
//
#include <hip/hip_runtime.h>

// ---------------------------------------------------------------------------
// BinaryTreeLSTM on MI455X (gfx1250): split-bf16 WMMA GEMMs + fused gates.
// Level kernels: h stored as split bf16 -> A staged via async global->LDS DMA.
// Leaf kernel: fp32 input staged with in-register hi/lo split.
// ---------------------------------------------------------------------------

#define B_TREES   32
#define N_LEAVES  512
#define IN_DIM    1024
#define MEM       1024
#define LEAF_ROWS (B_TREES * N_LEAVES)   // 16384

typedef __attribute__((ext_vector_type(16))) __bf16 v16bf;
typedef __attribute__((ext_vector_type(8)))  float  v8f;

union FragB {           // one 16x32 bf16 A-frag or 32x16 bf16 B-frag per lane
    v16bf v;
    uint4 q[2];
};

__device__ __forceinline__ unsigned short f2bf(float x) {
    unsigned u = __float_as_uint(x);
    unsigned r = (u + 0x7FFFu + ((u >> 16) & 1u)) >> 16;   // RNE
    return (unsigned short)r;
}
__device__ __forceinline__ float bf2f(unsigned short b) {
    return __uint_as_float(((unsigned)b) << 16);
}
__device__ __forceinline__ float sigm(float x) {
    return 1.0f / (1.0f + __expf(-x));
}
__device__ __forceinline__ void wait_async_all() {
    asm volatile("s_wait_asynccnt 0x0" ::: "memory");
}

// --------------------------- weight prep -----------------------------------
__global__ __launch_bounds__(256) void prep_split(const float* __restrict__ src,
                                                  unsigned short* __restrict__ hi,
                                                  unsigned short* __restrict__ lo,
                                                  long long n) {
    long long i = (long long)blockIdx.x * blockDim.x + threadIdx.x;
    long long stride = (long long)gridDim.x * blockDim.x;
    for (; i < n; i += stride) {
        float x = src[i];
        unsigned short h = f2bf(x);
        hi[i] = h;
        lo[i] = f2bf(x - bf2f(h));
    }
}

// --------------------------- fused GEMM + gates -----------------------------
// Tile: 128 rows x 32 cols, 8 waves (4 row-groups x 2 col-groups),
// per wave: 2 row-frags x 1 col-frag x NG gates, 3 WMMAs (split terms) each.
template <int K, int NG, bool IS_LEVEL>
__global__ __launch_bounds__(256) void treelstm_gemm(
    const float* __restrict__ Af32,          // leaf: [rows][K] fp32 input
    const unsigned short* __restrict__ AHi,  // level: [rows][K] bf16 hi
    const unsigned short* __restrict__ ALo,  // level: [rows][K] bf16 lo
    const unsigned short* __restrict__ Whi,  // [NG*MEM][K]
    const unsigned short* __restrict__ Wlo,
    const float* __restrict__ bias,          // b_fioux [4*MEM]
    const float* __restrict__ cprev,         // level: [rows][2*MEM]; leaf: null
    float* __restrict__ cout,                // [rows][MEM]
    unsigned short* __restrict__ hHiOut,     // [rows][MEM]
    unsigned short* __restrict__ hLoOut,     // [rows][MEM]
    float* __restrict__ hextra,              // final level -> d_out (f32)
    int rows) {
    __shared__ __align__(16) unsigned short lAhi[2][128][40];
    __shared__ __align__(16) unsigned short lAlo[2][128][40];

    const int tid  = threadIdx.x;
    const int lane = tid & 31;
    const int wid  = tid >> 5;
    const int wm   = wid & 3;    // row group: 32 rows each
    const int wn   = wid >> 2;   // col group: 16 cols each
    const int bm   = blockIdx.x * 128;
    const int bn   = blockIdx.y * 32;

    v8f acc[NG][2];
    const v8f vz = {0.f, 0.f, 0.f, 0.f, 0.f, 0.f, 0.f, 0.f};
#pragma unroll
    for (int g = 0; g < NG; ++g) {
        acc[g][0] = vz;
        acc[g][1] = vz;
    }

    // A staging: 256 threads cover 128 rows x 32 K; thread -> (row, 16-K half)
    const int srow = tid >> 1;
    const int skh  = (tid & 1) * 16;
    const bool srowOK = (bm + srow) < rows;
    const unsigned aBase = (unsigned)(bm + srow) * (unsigned)K + (unsigned)skh;

    const int r16 = lane & 15;
    const int kbA = (lane >> 4) * 16;   // A frag K base within chunk
    const int kbB = (lane >> 4) * 16;   // B frag K base within chunk

    // per-gate 32-bit element offsets into Whi/Wlo (span < 2^31)
    unsigned wOff[NG];
#pragma unroll
    for (int g = 0; g < NG; ++g)
        wOff[g] = (unsigned)(g * MEM + bn + wn * 16 + r16) * (unsigned)K + (unsigned)kbB;

    // ---------------- level: async DMA staging helper -----------------------
    auto stage_async = [&](int b, int kc) {
        if (srowOK) {
            const unsigned ldsHi = (unsigned)(unsigned long long)&lAhi[b][srow][skh];
            const unsigned ldsLo = (unsigned)(unsigned long long)&lAlo[b][srow][skh];
            const unsigned gOff  = (aBase + (unsigned)kc) * 2u;   // bytes
            asm volatile(
                "global_load_async_to_lds_b128 %0, %2, %3 offset:0\n\t"
                "global_load_async_to_lds_b128 %0, %2, %3 offset:16\n\t"
                "global_load_async_to_lds_b128 %1, %2, %4 offset:0\n\t"
                "global_load_async_to_lds_b128 %1, %2, %4 offset:16"
                :
                : "v"(ldsHi), "v"(ldsLo), "v"(gOff), "s"(AHi), "s"(ALo)
                : "memory");
        }
    };

    // ---------------- leaf: fp32 prefetch into registers ---------------------
    float xs[16];
    if constexpr (!IS_LEVEL) {
        if (srowOK) {
            const float4* p = (const float4*)(Af32 + aBase);
#pragma unroll
            for (int q = 0; q < 4; ++q) {
                float4 x = p[q];
                xs[4 * q + 0] = x.x; xs[4 * q + 1] = x.y;
                xs[4 * q + 2] = x.z; xs[4 * q + 3] = x.w;
            }
        } else {
#pragma unroll
            for (int e = 0; e < 16; ++e) xs[e] = 0.f;
        }
    } else {
        stage_async(0, 0);
        wait_async_all();
        __syncthreads();
    }

    int buf = 0;
    for (int kc = 0; kc < K; kc += 32, buf ^= 1) {
        if constexpr (!IS_LEVEL) {
            // ---- leaf: register -> LDS with hi/lo split ----
            unsigned hw[8], lw[8];
#pragma unroll
            for (int e = 0; e < 8; ++e) {
                float a = xs[2 * e], b = xs[2 * e + 1];
                unsigned short ha = f2bf(a), hb = f2bf(b);
                hw[e] = (unsigned)ha | ((unsigned)hb << 16);
                lw[e] = (unsigned)f2bf(a - bf2f(ha)) | ((unsigned)f2bf(b - bf2f(hb)) << 16);
            }
            *(uint4*)&lAhi[buf][srow][skh]     = make_uint4(hw[0], hw[1], hw[2], hw[3]);
            *(uint4*)&lAhi[buf][srow][skh + 8] = make_uint4(hw[4], hw[5], hw[6], hw[7]);
            *(uint4*)&lAlo[buf][srow][skh]     = make_uint4(lw[0], lw[1], lw[2], lw[3]);
            *(uint4*)&lAlo[buf][srow][skh + 8] = make_uint4(lw[4], lw[5], lw[6], lw[7]);
            if (kc + 32 < K) {
                if (srowOK) {
                    const float4* p = (const float4*)(Af32 + aBase + (unsigned)(kc + 32));
#pragma unroll
                    for (int q = 0; q < 4; ++q) {
                        float4 x = p[q];
                        xs[4 * q + 0] = x.x; xs[4 * q + 1] = x.y;
                        xs[4 * q + 2] = x.z; xs[4 * q + 3] = x.w;
                    }
                }
            }
            __syncthreads();
        } else {
            // ---- level: kick off DMA of NEXT chunk; overlaps this chunk's WMMAs
            if (kc + 32 < K) stage_async(buf ^ 1, kc + 32);
        }

        // ---- A fragments from LDS (contiguous K: k = 16*half + s, matches B) ----
        FragB ahi[2], alo[2];
#pragma unroll
        for (int rf = 0; rf < 2; ++rf) {
            const int rr = wm * 32 + rf * 16 + r16;
            ahi[rf].q[0] = *(const uint4*)&lAhi[buf][rr][kbA];
            ahi[rf].q[1] = *(const uint4*)&lAhi[buf][rr][kbA + 8];
            alo[rf].q[0] = *(const uint4*)&lAlo[buf][rr][kbA];
            alo[rf].q[1] = *(const uint4*)&lAlo[buf][rr][kbA + 8];
        }

        // ---- gate loop with B-fragment software prefetch ----
        FragB bhi, blo, bhiN, bloN;
        {
            const unsigned o = wOff[0] + (unsigned)kc;
            bhi.q[0] = ((const uint4*)(Whi + o))[0];
            bhi.q[1] = ((const uint4*)(Whi + o))[1];
            blo.q[0] = ((const uint4*)(Wlo + o))[0];
            blo.q[1] = ((const uint4*)(Wlo + o))[1];
        }
#pragma unroll
        for (int g = 0; g < NG; ++g) {
            if (g + 1 < NG) {
                const unsigned o = wOff[g + 1] + (unsigned)kc;
                bhiN.q[0] = ((const uint4*)(Whi + o))[0];
                bhiN.q[1] = ((const uint4*)(Whi + o))[1];
                bloN.q[0] = ((const uint4*)(Wlo + o))[0];
                bloN.q[1] = ((const uint4*)(Wlo + o))[1];
            }
#pragma unroll
            for (int rf = 0; rf < 2; ++rf) {
                acc[g][rf] = __builtin_amdgcn_wmma_f32_16x16x32_bf16(
                    false, ahi[rf].v, false, bhi.v, (short)0, acc[g][rf], false, false);
                acc[g][rf] = __builtin_amdgcn_wmma_f32_16x16x32_bf16(
                    false, alo[rf].v, false, bhi.v, (short)0, acc[g][rf], false, false);
                acc[g][rf] = __builtin_amdgcn_wmma_f32_16x16x32_bf16(
                    false, ahi[rf].v, false, blo.v, (short)0, acc[g][rf], false, false);
            }
            bhi = bhiN;
            blo = bloN;
        }

        if constexpr (IS_LEVEL) {
            wait_async_all();    // next chunk's DMA has landed
            __syncthreads();     // one barrier per chunk
        }
    }

    // ---- fused TreeLSTM cell epilogue ----
    const int m = bn + wn * 16 + (lane & 15);         // gate neuron index
    const int rbase = bm + wm * 32 + 8 * (lane >> 4); // C-frag M mapping
    const float bI = bias[MEM + m];
    const float bO = bias[2 * MEM + m];
    const float bU = bias[3 * MEM + m];
    const float bF = IS_LEVEL ? bias[m] : 0.f;

#pragma unroll
    for (int rf = 0; rf < 2; ++rf) {
#pragma unroll
        for (int v = 0; v < 8; ++v) {
            const int row = rbase + rf * 16 + v;
            if (row >= rows) continue;
            const float i = sigm(acc[0][rf][v] + bI);
            const float o = sigm(acc[1][rf][v] + bO);
            const float u = tanhf(acc[2][rf][v] + bU);
            float c;
            if constexpr (IS_LEVEL) {
                const float fl = sigm(acc[3][rf][v] + bF);
                const float fr = sigm(acc[4][rf][v] + bF);
                const unsigned cb = (unsigned)row * (unsigned)(2 * MEM) + (unsigned)m;
                const float cl = cprev[cb];
                const float cr = cprev[cb + MEM];
                c = i * u + fl * cl + fr * cr;
            } else {
                c = i * u;
            }
            const float h = o * tanhf(c);
            const unsigned idx = (unsigned)row * (unsigned)MEM + (unsigned)m;
            cout[idx] = c;
            const unsigned short hh = f2bf(h);
            hHiOut[idx] = hh;
            hLoOut[idx] = f2bf(h - bf2f(hh));
            if (IS_LEVEL && hextra != nullptr) hextra[idx] = h;
        }
    }
}

// --------------------------------- launch -----------------------------------
extern "C" void kernel_launch(void* const* d_in, const int* in_sizes, int n_in,
                              void* d_out, int out_size, void* d_ws, size_t ws_size,
                              hipStream_t stream) {
    (void)in_sizes; (void)n_in; (void)out_size; (void)ws_size;
    const float* X      = (const float*)d_in[0]; // [16384][1024]
    const float* wfioux = (const float*)d_in[1]; // [4096][1024]
    const float* bias   = (const float*)d_in[2]; // [4096]
    const float* wiouh  = (const float*)d_in[3]; // [3072][2048]
    const float* wfh    = (const float*)d_in[4]; // [2048][2048]

    char* ws = (char*)d_ws;
    size_t off = 0;
    auto take = [&](size_t bytes) -> char* {
        char* p = ws + off;
        off = (off + bytes + 255) & ~(size_t)255;
        return p;
    };
    unsigned short* WlHi = (unsigned short*)take((size_t)3 * MEM * IN_DIM * 2);
    unsigned short* WlLo = (unsigned short*)take((size_t)3 * MEM * IN_DIM * 2);
    unsigned short* WcHi = (unsigned short*)take((size_t)5 * MEM * 2 * MEM * 2);
    unsigned short* WcLo = (unsigned short*)take((size_t)5 * MEM * 2 * MEM * 2);
    float* cA = (float*)take((size_t)LEAF_ROWS * MEM * 4);
    float* cB = (float*)take((size_t)(LEAF_ROWS / 2) * MEM * 4);
    unsigned short* hAHi = (unsigned short*)take((size_t)LEAF_ROWS * MEM * 2);
    unsigned short* hALo = (unsigned short*)take((size_t)LEAF_ROWS * MEM * 2);
    unsigned short* hBHi = (unsigned short*)take((size_t)(LEAF_ROWS / 2) * MEM * 2);
    unsigned short* hBLo = (unsigned short*)take((size_t)(LEAF_ROWS / 2) * MEM * 2);

    // ---- pre-split weights into bf16 hi/lo (one-time, then L2-resident) ----
    prep_split<<<1024, 256, 0, stream>>>(wfioux + (size_t)MEM * IN_DIM,
                                         WlHi, WlLo, (long long)3 * MEM * IN_DIM);
    prep_split<<<1024, 256, 0, stream>>>(wiouh, WcHi, WcLo,
                                         (long long)3 * MEM * 2 * MEM);
    prep_split<<<1024, 256, 0, stream>>>(wfh,
                                         WcHi + (size_t)3 * MEM * 2 * MEM,
                                         WcLo + (size_t)3 * MEM * 2 * MEM,
                                         (long long)2 * MEM * 2 * MEM);

    // ---- leaf level: [16384 x 1024] @ W^T, 3 gates ----
    {
        dim3 grid(LEAF_ROWS / 128, MEM / 32);
        treelstm_gemm<IN_DIM, 3, false><<<grid, 256, 0, stream>>>(
            X, nullptr, nullptr, WlHi, WlLo, bias, nullptr,
            cA, hAHi, hALo, nullptr, LEAF_ROWS);
    }

    // ---- 9 internal levels: child_h = flat previous-level (hHi, hLo) ----
    float* cSrc = cA; float* cDst = cB;
    unsigned short* hSrcHi = hAHi; unsigned short* hSrcLo = hALo;
    unsigned short* hDstHi = hBHi; unsigned short* hDstLo = hBLo;
    int rows = LEAF_ROWS;
    for (int lvl = 1; lvl <= 9; ++lvl) {
        rows >>= 1;
        float* hextra = (lvl == 9) ? (float*)d_out : nullptr;
        dim3 grid((rows + 127) / 128, MEM / 32);
        treelstm_gemm<2 * MEM, 5, true><<<grid, 256, 0, stream>>>(
            nullptr, hSrcHi, hSrcLo, WcHi, WcLo, bias, cSrc,
            cDst, hDstHi, hDstLo, hextra, rows);
        float* tf;
        tf = cSrc; cSrc = cDst; cDst = tf;
        unsigned short* ts;
        ts = hSrcHi; hSrcHi = hDstHi; hDstHi = ts;
        ts = hSrcLo; hSrcLo = hDstLo; hDstLo = ts;
    }
}
